// GNNLayer_70153995813128
// MI455X (gfx1250) — compile-verified, matrix operand order checked by reading, not run on hardware
//
#include <hip/hip_runtime.h>
#include <hip/hip_bf16.h>
#include <math.h>

// ---------------------------------------------------------------------------
// GNN layer, MI455X (gfx1250, wave32). bf16 WMMA GEMMs w/ fp32 accumulate,
// TDM (tensor_load_to_lds) double-buffered weight staging, fused
// LayerNorm/ReLU/softmax epilogues.  B=64, N=27, C=512.
// ---------------------------------------------------------------------------

typedef __bf16 bf16_t;
typedef __attribute__((ext_vector_type(16))) __bf16 v16bf;
typedef __attribute__((ext_vector_type(8)))  float  v8f;
typedef __attribute__((ext_vector_type(4)))  unsigned int v4u;
typedef __attribute__((ext_vector_type(8)))  int    v8i;
typedef __attribute__((ext_vector_type(4)))  int    v4i;

#define BATCH 64
#define NNODE 27
#define CH    512
#define EDGES 729                 // 27*27 per batch
#define ROWS_X (BATCH * NNODE)    // 1728
#define ROWS_E (BATCH * EDGES)    // 46656

// ---------------------------------------------------------------------------
// Fragment loader for 16-bit A/B matrices of V_WMMA_F32_16X16X32_BF16.
// ISA 7.12.2 layout: lane l holds row (l&15); element e maps to
// k = koff + (e>>3)*16 + (e&7) + 8*(l>>4).  Contiguous groups of 8 K values
// merge into ds_load_b128.
// ---------------------------------------------------------------------------
__device__ __forceinline__ v16bf load_frag(const bf16_t* __restrict__ base,
                                           int row, int row_stride,
                                           int koff, int hk) {
  const bf16_t* p = base + row * row_stride + koff + hk;
  v16bf f;
#pragma unroll
  for (int g = 0; g < 2; ++g)
#pragma unroll
    for (int i = 0; i < 8; ++i)
      f[g * 8 + i] = p[g * 16 + i];
  return f;
}

// ---------------------------------------------------------------------------
// TDM: async-copy a 2D bf16 tile (32 contiguous K x 128 N-rows, row stride
// 512 elements) from global into LDS.  D# per ISA ch.8:
//   group0: count=1 | lds_addr | global_addr | type=2
//   group1: data_size=2B, tensor_dim0/1=512, tile_dim0=32, tile_dim1=128,
//           tensor_dim0_stride=512
// Tracked by TENSORcnt; waited with s_wait_tensorcnt.
// ---------------------------------------------------------------------------
__device__ __forceinline__ void tdm_load_wslice(const bf16_t* g, bf16_t* lds) {
  unsigned long long ga = (unsigned long long)(uintptr_t)g;
  unsigned lo = (unsigned)(uintptr_t)lds;   // addr[31:0] == LDS byte offset
  v4u g0 = { 1u,                                       // count=1 (valid, user)
             lo,                                       // lds_addr
             (unsigned)ga,                             // global_addr[31:0]
             (unsigned)((ga >> 32) & 0x1ffffffu) | (2u << 30) };  // [56:32]|type=2
  v8i g1 = { (int)(1u << 16),                          // data_size = 2 bytes
             (int)((512u & 0xffffu) << 16),            // tensor_dim0 lo16
             (int)(512u << 16),                        // dim0 hi=0 | tensor_dim1 lo16
             (int)(32u << 16),                         // dim1 hi=0 | tile_dim0=32
             (int)128,                                 // tile_dim1=128, tile_dim2=0
             (int)512,                                 // tensor_dim0_stride lo32
             0, 0 };                                   // stride hi / dim1_stride
  v4i gz = { 0, 0, 0, 0 };
#if defined(__clang_major__) && (__clang_major__ >= 23)
  v8i g4 = { 0, 0, 0, 0, 0, 0, 0, 0 };
  __builtin_amdgcn_tensor_load_to_lds(g0, g1, gz, gz, g4, 0);
#else
  __builtin_amdgcn_tensor_load_to_lds(g0, g1, gz, gz, 0);
#endif
}

// ---------------------------------------------------------------------------
// K0: convert x -> bf16 (row major) and the 5 weights -> bf16 TRANSPOSED
// (n-major) so B-fragments read contiguous K.
// wt layout order: WA(0) WB(1) WE(2) WU(3) WV(4)
// ---------------------------------------------------------------------------
__global__ void k_convert(const float* __restrict__ x,
                          const float* __restrict__ WA,
                          const float* __restrict__ WB,
                          const float* __restrict__ WE,
                          const float* __restrict__ WU,
                          const float* __restrict__ WV,
                          bf16_t* __restrict__ xb,
                          bf16_t* __restrict__ wt_all) {
  const int seg = blockIdx.y;
  const int idx = blockIdx.x * 256 + threadIdx.x;
  if (seg == 0) {
    if (idx < ROWS_X * CH) xb[idx] = (bf16_t)x[idx];
  } else {
    if (idx < CH * CH) {
      const float* W = (seg == 1) ? WA : (seg == 2) ? WB
                     : (seg == 3) ? WE : (seg == 4) ? WU : WV;
      const int k = idx >> 9;        // source row
      const int n = idx & 511;       // source col
      wt_all[(size_t)(seg - 1) * (CH * CH) + (size_t)n * CH + k] = (bf16_t)W[idx];
    }
  }
}

// ---------------------------------------------------------------------------
// K1: node projections.  out[z] = x @ W[z] for z in {WA,WB,WV,WU}.
// Block = 128 threads (4 waves).  M-tile = 16 rows, full 512 output cols.
// Each wave owns output cols [wv*128, wv*128+128) and TDM-stages only its own
// 128-row W slice -> no block barrier inside the K loop, double-buffered.
// ---------------------------------------------------------------------------
__global__ __launch_bounds__(128) void k_node_gemm(
    const bf16_t* __restrict__ xb,      // [1728][512]
    const bf16_t* __restrict__ wt_all,  // 5 x [512][512] (transposed)
    float* __restrict__ out_all)        // 4 x [1728][512]
{
  __shared__ __align__(16) bf16_t Abuf[16 * CH];          // 16 KB
  __shared__ __align__(16) bf16_t Wbuf[2][4][128 * 32];   // 64 KB (2 bufs x 4 waves)

  const int tid  = threadIdx.x;
  const int lane = tid & 31;
  const int wv   = tid >> 5;
  const int hk   = (lane >> 4) * 8;
  const int lo16 = lane & 15;
  const int row0 = blockIdx.x * 16;
  const int z    = blockIdx.z;
  // z: 0->WA(Vix) 1->WB(Vjx) 2->WV(Ujx) 3->WU(XU); wt slots A,B,E,U,V
  const int wslot = (z == 2) ? 4 : ((z == 3) ? 3 : z);
  const bf16_t* wt = wt_all + (size_t)wslot * (CH * CH);
  float* out = out_all + (size_t)z * (ROWS_X * CH);

  const bf16_t* wt_w = wt + (size_t)(wv * 128) * CH;   // this wave's N rows
  bf16_t* wb0 = &Wbuf[0][wv][0];
  bf16_t* wb1 = &Wbuf[1][wv][0];

  // kick off first two TDM W-slice loads (overlap with A staging)
  tdm_load_wslice(wt_w + 0,  wb0);
  tdm_load_wslice(wt_w + 32, wb1);

  // stage A tile (rows contiguous in memory)
  {
    const uint4* src = reinterpret_cast<const uint4*>(xb + (size_t)row0 * CH);
    uint4* dst = reinterpret_cast<uint4*>(Abuf);
    for (int i = tid; i < 16 * CH / 8; i += 128) dst[i] = src[i];
  }
  __syncthreads();

  v8f acc[8] = {};

  auto compute_step = [&](const bf16_t* wslice, int kb) {
    v16bf a = load_frag(Abuf, lo16, CH, kb, hk);
#pragma unroll
    for (int t = 0; t < 8; ++t) {
      v16bf bm = load_frag(wslice, t * 16 + lo16, 32, 0, hk);
      acc[t] = __builtin_amdgcn_wmma_f32_16x16x32_bf16(
          false, a, false, bm, (short)0, acc[t], false, false);
    }
  };

  for (int it = 0; it < 14; ++it) {
    __builtin_amdgcn_s_wait_tensorcnt(1);      // oldest in-flight slice ready
    compute_step((it & 1) ? wb1 : wb0, it * 32);
    tdm_load_wslice(wt_w + (it + 2) * 32, (it & 1) ? wb1 : wb0);
  }
  __builtin_amdgcn_s_wait_tensorcnt(1);
  compute_step(wb0, 14 * 32);
  __builtin_amdgcn_s_wait_tensorcnt(0);
  compute_step(wb1, 15 * 32);

#pragma unroll
  for (int t = 0; t < 8; ++t) {
    const int n = wv * 128 + t * 16 + lo16;
#pragma unroll
    for (int r = 0; r < 8; ++r) {
      const int m = r + hk;                 // C/D layout: row = r + 8*(lane>>4)
      out[(size_t)(row0 + m) * CH + n] = acc[t][r];
    }
  }
}

// ---------------------------------------------------------------------------
// K2: fused edge path.
//   e_lin = edge @ WE ; msg = Vix[i] + Vjx[j]
//   edge_out = edge + relu(LN(msg + e_lin) * ge + be)
// Same TDM-pipelined GEMM; row LayerNorm via LDS float atomics.
// ---------------------------------------------------------------------------
__global__ __launch_bounds__(128) void k_edge_fused(
    const float* __restrict__ edge,     // [46656][512]
    const bf16_t* __restrict__ wtE,     // [512][512] transposed
    const float* __restrict__ Vix,      // [1728][512]
    const float* __restrict__ Vjx,      // [1728][512]
    const float* __restrict__ ge,
    const float* __restrict__ be,
    float* __restrict__ edge_out)       // [46656][512]
{
  __shared__ __align__(16) bf16_t Abuf[16 * CH];
  __shared__ __align__(16) bf16_t Wbuf[2][4][128 * 32];
  __shared__ float s_sum[16];
  __shared__ float s_ssq[16];

  const int tid  = threadIdx.x;
  const int lane = tid & 31;
  const int wv   = tid >> 5;
  const int hk   = (lane >> 4) * 8;
  const int lo16 = lane & 15;
  const int row0 = blockIdx.x * 16;     // 46656 / 16 = 2916 blocks exactly

  const bf16_t* wt_w = wtE + (size_t)(wv * 128) * CH;
  bf16_t* wb0 = &Wbuf[0][wv][0];
  bf16_t* wb1 = &Wbuf[1][wv][0];

  tdm_load_wslice(wt_w + 0,  wb0);
  tdm_load_wslice(wt_w + 32, wb1);

  if (tid < 16) { s_sum[tid] = 0.f; s_ssq[tid] = 0.f; }

  // prefetch next block's A tile (global_prefetch_b8)
  if (tid == 0 && row0 + 16 < ROWS_E)
    __builtin_prefetch(edge + (size_t)(row0 + 16) * CH, 0, 1);

  // stage A tile: convert fp32 edge rows -> bf16
  {
    const float4* src = reinterpret_cast<const float4*>(edge + (size_t)row0 * CH);
    for (int i = tid; i < 16 * CH / 4; i += 128) {
      float4 v = src[i];
      Abuf[i * 4 + 0] = (bf16_t)v.x;
      Abuf[i * 4 + 1] = (bf16_t)v.y;
      Abuf[i * 4 + 2] = (bf16_t)v.z;
      Abuf[i * 4 + 3] = (bf16_t)v.w;
    }
  }
  __syncthreads();

  v8f acc[8] = {};

  auto compute_step = [&](const bf16_t* wslice, int kb) {
    v16bf a = load_frag(Abuf, lo16, CH, kb, hk);
#pragma unroll
    for (int t = 0; t < 8; ++t) {
      v16bf bm = load_frag(wslice, t * 16 + lo16, 32, 0, hk);
      acc[t] = __builtin_amdgcn_wmma_f32_16x16x32_bf16(
          false, a, false, bm, (short)0, acc[t], false, false);
    }
  };

  for (int it = 0; it < 14; ++it) {
    __builtin_amdgcn_s_wait_tensorcnt(1);
    compute_step((it & 1) ? wb1 : wb0, it * 32);
    tdm_load_wslice(wt_w + (it + 2) * 32, (it & 1) ? wb1 : wb0);
  }
  __builtin_amdgcn_s_wait_tensorcnt(1);
  compute_step(wb0, 14 * 32);
  __builtin_amdgcn_s_wait_tensorcnt(0);
  compute_step(wb1, 15 * 32);

  // ---- epilogue: msg add + per-row mean/var partials ----
  float ps[8] = {0.f, 0.f, 0.f, 0.f, 0.f, 0.f, 0.f, 0.f};
  float pq[8] = {0.f, 0.f, 0.f, 0.f, 0.f, 0.f, 0.f, 0.f};
#pragma unroll
  for (int t = 0; t < 8; ++t) {
    const int n = wv * 128 + t * 16 + lo16;
#pragma unroll
    for (int r = 0; r < 8; ++r) {
      const int gr = row0 + hk + r;          // global edge row
      const int b  = gr / EDGES;
      const int e  = gr - b * EDGES;
      const int i  = e / NNODE;
      const int j  = e - i * NNODE;
      float v = acc[t][r]
              + Vix[(size_t)(b * NNODE + i) * CH + n]
              + Vjx[(size_t)(b * NNODE + j) * CH + n];
      acc[t][r] = v;
      ps[r] += v;
      pq[r] += v * v;
    }
  }
#pragma unroll
  for (int r = 0; r < 8; ++r) {
    atomicAdd(&s_sum[hk + r], ps[r]);
    atomicAdd(&s_ssq[hk + r], pq[r]);
  }
  __syncthreads();

#pragma unroll
  for (int t = 0; t < 8; ++t) {
    const int n = wv * 128 + t * 16 + lo16;
    const float gn = ge[n], bn = be[n];
#pragma unroll
    for (int r = 0; r < 8; ++r) {
      const int gr = row0 + hk + r;
      const float mean = s_sum[hk + r] * (1.f / (float)CH);
      const float var  = s_ssq[hk + r] * (1.f / (float)CH) - mean * mean;
      const float rstd = rsqrtf(var + 1e-5f);
      float y = (acc[t][r] - mean) * rstd * gn + bn;
      y = fmaxf(y, 0.f);
      edge_out[(size_t)gr * CH + n] = edge[(size_t)gr * CH + n] + y;
    }
  }
}

// ---------------------------------------------------------------------------
// K3: gate + aggregate + node output.
//   p = softmax_j(sigmoid(edge_out[b,i,j,:]))
//   agg = sum_j p * Ujx[b,j,:] / 27 ; x_new = XU + agg
//   x_out = relu(x + LN(x_new) * gv + bv)
// One block per (b, i); 128 threads x 4 channels each.
// ---------------------------------------------------------------------------
__global__ __launch_bounds__(128) void k_gate_agg(
    const float* __restrict__ x,         // [1728][512]
    const float* __restrict__ edge_out,  // [46656][512]
    const float* __restrict__ Ujx,       // [1728][512]
    const float* __restrict__ XU,        // [1728][512]
    const float* __restrict__ gv,
    const float* __restrict__ bv,
    float* __restrict__ x_out)           // [1728][512]
{
  __shared__ float s_sum;
  __shared__ float s_ssq;

  const int tid = threadIdx.x;
  const int bi  = blockIdx.x;            // b*27 + i
  const int b   = bi / NNODE;
  const int i   = bi - b * NNODE;

  if (tid == 0) { s_sum = 0.f; s_ssq = 0.f; }
  __syncthreads();

  float xn[4];
  float lsum = 0.f, lssq = 0.f;
#pragma unroll
  for (int q = 0; q < 4; ++q) {
    const int c = tid + q * 128;
    float num = 0.f, den = 0.f;
    for (int j = 0; j < NNODE; ++j) {
      const float eo = edge_out[((size_t)b * EDGES + i * NNODE + j) * CH + c];
      const float s  = 1.f / (1.f + __expf(-eo));   // sigmoid
      const float w  = __expf(s);                   // softmax weight (s in (0,1): safe)
      den += w;
      num += w * Ujx[(size_t)(b * NNODE + j) * CH + c];
    }
    const float agg = (num / den) * (1.f / (float)NNODE);
    const float v   = XU[(size_t)bi * CH + c] + agg;
    xn[q] = v;
    lsum += v;
    lssq += v * v;
  }
  atomicAdd(&s_sum, lsum);
  atomicAdd(&s_ssq, lssq);
  __syncthreads();

  const float mean = s_sum * (1.f / (float)CH);
  const float var  = s_ssq * (1.f / (float)CH) - mean * mean;
  const float rstd = rsqrtf(var + 1e-5f);
#pragma unroll
  for (int q = 0; q < 4; ++q) {
    const int c = tid + q * 128;
    const float ln = (xn[q] - mean) * rstd * gv[c] + bv[c];
    x_out[(size_t)bi * CH + c] = fmaxf(x[(size_t)bi * CH + c] + ln, 0.f);
  }
}

// ---------------------------------------------------------------------------
extern "C" void kernel_launch(void* const* d_in, const int* in_sizes, int n_in,
                              void* d_out, int out_size, void* d_ws, size_t ws_size,
                              hipStream_t stream) {
  const float* x    = (const float*)d_in[0];
  const float* edge = (const float*)d_in[1];
  const float* WA   = (const float*)d_in[2];
  const float* WB   = (const float*)d_in[3];
  const float* WE   = (const float*)d_in[4];
  const float* WU   = (const float*)d_in[5];
  const float* WV   = (const float*)d_in[6];
  const float* gv   = (const float*)d_in[7];
  const float* bv   = (const float*)d_in[8];
  const float* ge   = (const float*)d_in[9];
  const float* be   = (const float*)d_in[10];

  float* x_out    = (float*)d_out;                       // 1728*512
  float* edge_out = (float*)d_out + (size_t)ROWS_X * CH; // 46656*512

  // workspace layout (16B aligned offsets):
  //   [0)        x_bf16 : 884736  * 2 B = 1769472
  //   [1769472)  wt_all : 5*262144* 2 B = 2621440  (A,B,E,U,V transposed)
  //   [4390912)  proj   : 4*884736* 4 B            (Vix,Vjx,Ujx,XU)
  char* ws = (char*)d_ws;
  bf16_t* xb     = (bf16_t*)ws;
  bf16_t* wt_all = (bf16_t*)(ws + 1769472);
  float*  proj   = (float*)(ws + 1769472 + 2621440);
  float*  Vix = proj;
  float*  Vjx = proj + (size_t)1 * ROWS_X * CH;
  float*  Ujx = proj + (size_t)2 * ROWS_X * CH;
  float*  XU  = proj + (size_t)3 * ROWS_X * CH;

  // K0: precision conversion (+ weight transpose)
  k_convert<<<dim3((ROWS_X * CH + 255) / 256, 6, 1), 256, 0, stream>>>(
      x, WA, WB, WE, WU, WV, xb, wt_all);

  // K1: four node projections (z selects weight/output)
  k_node_gemm<<<dim3(ROWS_X / 16, 1, 4), 128, 0, stream>>>(xb, wt_all, proj);

  // K2: fused edge GEMM + msg + LayerNorm + ReLU + residual
  k_edge_fused<<<dim3(ROWS_E / 16, 1, 1), 128, 0, stream>>>(
      edge, wt_all + (size_t)2 * CH * CH, Vix, Vjx, ge, be, edge_out);

  // K3: gate/softmax aggregation + node LayerNorm + residual ReLU
  k_gate_agg<<<dim3(ROWS_X, 1, 1), 128, 0, stream>>>(
      x, edge_out, Ujx, XU, gv, bv, x_out);
}
